// FBMambaBlock_69449621176664
// MI455X (gfx1250) — matrix operand order repaired and checked by reading, not compiled, not run on hardware
//
#include <hip/hip_runtime.h>
#include <hip/hip_bf16.h>
#include <cstdint>

// ---------------- problem constants ----------------
#define BATCH    8
#define SEQL     1024
#define DMODEL   512
#define DINNER   1024
#define DSTATE   64
#define HEADD    64
#define NHEADS   16
#define DCONV    4
#define CONVCH   (DINNER + 2 * DSTATE)           // 1152
#define DPROJ    (2 * DINNER + 2 * DSTATE + NHEADS) // 2192
#define NTOK     (BATCH * SEQL)                  // 8192
#define NPAD_IN  2240                            // DPROJ rounded up to 64

typedef __attribute__((ext_vector_type(16))) __bf16 v16bf;
typedef __attribute__((ext_vector_type(8)))  float  v8f;

// pack two floats into one dword holding two bf16 (RNE)
__device__ __forceinline__ unsigned int pack2bf(float lo, float hi) {
    unsigned int a = __builtin_bit_cast(unsigned int, lo);
    unsigned int b = __builtin_bit_cast(unsigned int, hi);
    a += 0x7FFFu + ((a >> 16) & 1u);
    b += 0x7FFFu + ((b >> 16) & 1u);
    return (a >> 16) | (b & 0xFFFF0000u);
}

union Frag16 { uint4 q[2]; v16bf v; };

template <int PAT>
__device__ __forceinline__ float swz_add(float v) {
    int t = __builtin_amdgcn_ds_swizzle(__builtin_bit_cast(int, v), PAT);
    return v + __builtin_bit_cast(float, t);
}
__device__ __forceinline__ float wave_sum(float v) {
    v = swz_add<0x041F>(v);
    v = swz_add<0x081F>(v);
    v = swz_add<0x101F>(v);
    v = swz_add<0x201F>(v);
    v = swz_add<0x401F>(v);
    return v;
}

__device__ __forceinline__ float silu(float x) { return x / (1.0f + __expf(-x)); }

// ======================================================================
// elementwise f32 -> packed bf16 (8 floats -> uint4 per thread)
// ======================================================================
__global__ __launch_bounds__(256)
void cvt_f32_bf16_kernel(const float* __restrict__ src, uint4* __restrict__ dst, int n4)
{
    int i = blockIdx.x * 256 + threadIdx.x;
    if (i >= n4) return;
    const float4* s = (const float4*)src + (size_t)i * 2;
    float4 f0 = s[0], f1 = s[1];
    uint4 w;
    w.x = pack2bf(f0.x, f0.y); w.y = pack2bf(f0.z, f0.w);
    w.z = pack2bf(f1.x, f1.y); w.w = pack2bf(f1.z, f1.w);
    dst[i] = w;
}

// ======================================================================
// weight transpose + convert: W[K][N] f32 -> out[Npad][K] packed bf16 pairs.
// rows n >= N are zero-filled so GEMM async loads need no predication.
// ======================================================================
__global__ __launch_bounds__(256)
void transpose_cvt_kernel(const float* __restrict__ W, unsigned* __restrict__ out,
                          int N, int Npad, int K)
{
    int idx = blockIdx.x * 256 + threadIdx.x;
    int kh = K >> 1;
    if (idx >= Npad * kh) return;
    int n  = idx / kh;
    int kp = idx % kh;
    float a = 0.0f, b = 0.0f;
    if (n < N) {
        a = W[(size_t)(2 * kp)     * N + n];
        b = W[(size_t)(2 * kp + 1) * N + n];
    }
    out[(size_t)n * kh + kp] = pack2bf(a, b);
}

// ======================================================================
// bf16 WMMA GEMM with async-to-LDS staging and double buffering.
//   C[M,N](f32) = A[M,K](bf16 row-major) * Bt[N,K](bf16, pre-transposed)
// Block tile 64x64, k-step 32; 8 waves, each computes a 16x32 output tile
// with two v_wmma_f32_16x16x32_bf16 per k-step. Tiles staged with
// global_load_async_to_lds_b128 (ASYNCcnt); last iteration peeled so the
// steady-state loop body is branch-free.
// ======================================================================
#define BLK_M 64
#define BLK_N 64
#define BLK_K 32

__global__ __launch_bounds__(256)
void gemm_bf16_async(const __bf16* __restrict__ A, int lda,
                     const __bf16* __restrict__ Bt, int ldb,
                     float* __restrict__ C, int ldc,
                     int N, int K,
                     const float* __restrict__ resid, int res_ld)
{
    __shared__ unsigned int As32[2][BLK_M][BLK_K / 2];   // 2 x 4KB
    __shared__ unsigned int Bs32[2][BLK_N][BLK_K / 2];   // 2 x 4KB

    const int tid  = threadIdx.x;
    const int lane = tid & 31;
    const int wave = tid >> 5;       // 0..7
    const int wm   = wave >> 1;      // 0..3
    const int wn   = wave & 1;       // 0..1
    const int half = lane >> 4;
    const int l16  = lane & 15;

    const int m0 = blockIdx.x * BLK_M;
    const int n0 = blockIdx.y * BLK_N;

    // staging: thread owns one 16B segment of one row, both operands
    const int srow = tid >> 2;             // 0..63
    const int sseg = (tid & 3) * 4;        // dword offset 0,4,8,12

    const __bf16* gA = A  + (size_t)(m0 + srow) * lda + sseg * 2;
    const __bf16* gB = Bt + (size_t)(n0 + srow) * ldb + sseg * 2;

    v8f acc0 = {}; v8f acc1 = {};
    const int ktiles = K / BLK_K;          // >= 16 for all uses

    auto issue = [&](int kt, int buf) {
        unsigned la = (unsigned)(uintptr_t)&As32[buf][srow][sseg];
        unsigned lb = (unsigned)(uintptr_t)&Bs32[buf][srow][sseg];
        unsigned long long ga = (unsigned long long)(uintptr_t)(gA + kt * BLK_K);
        unsigned long long gb = (unsigned long long)(uintptr_t)(gB + kt * BLK_K);
        asm volatile("global_load_async_to_lds_b128 %0, %1, off"
                     :: "v"(la), "v"(ga) : "memory");
        asm volatile("global_load_async_to_lds_b128 %0, %1, off"
                     :: "v"(lb), "v"(gb) : "memory");
    };

    auto compute = [&](int buf) {
        // A fragment (16x32 bf16): lane<16 K0..15; lane>=16 K16..31
        Frag16 fa;
        {
            const unsigned int* row = &As32[buf][wm * 16 + l16][0];
            fa.q[0] = *(const uint4*)(row + half * 8);
            fa.q[1] = *(const uint4*)(row + half * 8 + 4);
        }
        // B fragments (32x16): lane<16 K0..15 of col l16; lane>=16 K16..31
        Frag16 fb0, fb1;
        {
            const unsigned int* r0 = &Bs32[buf][wn * 32 + l16][0];
            const unsigned int* r1 = &Bs32[buf][wn * 32 + l16 + 16][0];
            fb0.q[0] = *(const uint4*)(r0 + half * 8);
            fb0.q[1] = *(const uint4*)(r0 + half * 8 + 4);
            fb1.q[0] = *(const uint4*)(r1 + half * 8);
            fb1.q[1] = *(const uint4*)(r1 + half * 8 + 4);
        }
        acc0 = __builtin_amdgcn_wmma_f32_16x16x32_bf16(false, fa.v, false, fb0.v,
                                                       (short)0, acc0, false, false);
        acc1 = __builtin_amdgcn_wmma_f32_16x16x32_bf16(false, fa.v, false, fb1.v,
                                                       (short)0, acc1, false, false);
    };

    issue(0, 0);
    // steady state: branch-free body, next tile in flight during compute
    for (int i = 0; i < ktiles - 1; ++i) {
        const int cur = i & 1;
        issue(i + 1, cur ^ 1);                            // prefetch next tile
        asm volatile("s_wait_asynccnt 0x2" ::: "memory"); // tile i landed
        __syncthreads();
        compute(cur);
        __syncthreads();
    }
    // last tile
    asm volatile("s_wait_asynccnt 0x0" ::: "memory");
    __syncthreads();
    compute((ktiles - 1) & 1);

    // store C/D layout: VGPR r -> row (half*8 + r), col l16
    {
        int rowb = m0 + wm * 16 + half * 8;
        int col0 = n0 + wn * 32 + l16;
        int col1 = col0 + 16;
        #pragma unroll
        for (int r = 0; r < 8; ++r) {
            int rr = rowb + r;
            if (col0 < N) {
                float v = acc0[r];
                if (resid) v += resid[(size_t)rr * res_ld + col0];
                C[(size_t)rr * ldc + col0] = v;
            }
            if (col1 < N) {
                float v = acc1[r];
                if (resid) v += resid[(size_t)rr * res_ld + col1];
                C[(size_t)rr * ldc + col1] = v;
            }
        }
    }
}

// ======================================================================
// depthwise causal conv1d (k=4) + bias + SiLU over the xBC slice of zxbcdt.
// dir=0: taps t-3..t.  dir=1: out[t] = b + sum_k w[k]*x[t+3-k].
// ======================================================================
__global__ __launch_bounds__(256)
void conv_silu_kernel(const float* __restrict__ zxbcdt,
                      const float* __restrict__ convw,
                      const float* __restrict__ convb,
                      float* __restrict__ xbc, int dir)
{
    int idx = blockIdx.x * blockDim.x + threadIdx.x;
    if (idx >= BATCH * SEQL * CONVCH) return;
    int ch = idx % CONVCH;
    int t  = (idx / CONVCH) % SEQL;
    int b  =  idx / (CONVCH * SEQL);

    const float* xcol = zxbcdt + (size_t)b * SEQL * DPROJ + (DINNER + ch);
    float acc = convb[ch];
    if (dir == 0) {
        #pragma unroll
        for (int k = 0; k < DCONV; ++k) {
            int tt = t - (DCONV - 1) + k;
            if (tt >= 0) acc += convw[ch * DCONV + k] * xcol[(size_t)tt * DPROJ];
        }
    } else {
        #pragma unroll
        for (int k = 0; k < DCONV; ++k) {
            int tt = t + (DCONV - 1) - k;
            if (tt < SEQL) acc += convw[ch * DCONV + k] * xcol[(size_t)tt * DPROJ];
        }
    }
    xbc[((size_t)b * SEQL + t) * CONVCH + ch] = silu(acc);
}

// ======================================================================
// selective scan: one workgroup per (batch, head); 64x64 fp32 state lives in
// registers (16/thread). thread = (p = t>>2, c = t&3); 4-lane ds_swizzle y-reduce.
// ======================================================================
__global__ __launch_bounds__(256)
void ssm_scan_kernel(const float* __restrict__ zxbcdt,
                     const float* __restrict__ xbc,
                     const float* __restrict__ dt_bias,
                     const float* __restrict__ A_log,
                     const float* __restrict__ Dparam,
                     float* __restrict__ y, int dir)
{
    const int b = blockIdx.x / NHEADS;
    const int h = blockIdx.x % NHEADS;
    const int tid = threadIdx.x;
    const int p  = tid >> 2;
    const int c  = tid & 3;
    const int n0 = c * 16;

    __shared__ float sx[HEADD], sB[DSTATE], sC[DSTATE];
    __shared__ float sdt, sdA;

    const float A     = -__expf(A_log[h]);
    const float dbias = dt_bias[h];
    const float Dh    = Dparam[h];

    float hst[16];
    #pragma unroll
    for (int j = 0; j < 16; ++j) hst[j] = 0.0f;

    for (int step = 0; step < SEQL; ++step) {
        const int t = dir ? (SEQL - 1 - step) : step;
        const float* xrow = xbc + ((size_t)b * SEQL + t) * CONVCH;
        if (tid < 64)        sx[tid]        = xrow[h * HEADD + tid];
        else if (tid < 128)  sB[tid - 64]   = xrow[DINNER + (tid - 64)];
        else if (tid < 192)  sC[tid - 128]  = xrow[DINNER + DSTATE + (tid - 128)];
        else if (tid == 192) {
            float draw = zxbcdt[((size_t)b * SEQL + t) * DPROJ + (DINNER + CONVCH) + h] + dbias;
            float dtv  = (draw > 20.0f) ? draw : __logf(1.0f + __expf(draw));
            sdt = dtv;
            sdA = __expf(dtv * A);
        }
        __syncthreads();

        const float dtv = sdt, dA = sdA;
        const float xp   = sx[p];
        const float coef = dtv * xp;
        float yp = 0.0f;
        #pragma unroll
        for (int j = 0; j < 16; ++j) {
            float hv = fmaf(hst[j], dA, coef * sB[n0 + j]);
            hst[j] = hv;
            yp = fmaf(hv, sC[n0 + j], yp);
        }
        yp = swz_add<0x041F>(yp);   // xor 1
        yp = swz_add<0x081F>(yp);   // xor 2
        if (c == 0)
            y[((size_t)b * SEQL + t) * DINNER + h * HEADD + p] = yp + Dh * xp;
        __syncthreads();
    }
}

// ======================================================================
// ybf = bf16( RMSNorm(y * silu(z)) * norm_w ). one workgroup per token,
// 4 consecutive elements per thread -> packed dword stores.
// ======================================================================
__global__ __launch_bounds__(256)
void gate_rmsnorm_bf16_kernel(const float* __restrict__ zxbcdt,
                              const float* __restrict__ norm_w,
                              const float* __restrict__ yf,
                              __bf16* __restrict__ ybf)
{
    const int token = blockIdx.x;
    const float* zrow = zxbcdt + (size_t)token * DPROJ;   // z at offset 0
    const float* yrow = yf + (size_t)token * DINNER;

    __shared__ float red[8];
    const int i0 = threadIdx.x * 4;
    float4 z4 = *(const float4*)(zrow + i0);
    float4 y4 = *(const float4*)(yrow + i0);
    float4 v;
    v.x = y4.x * silu(z4.x);
    v.y = y4.y * silu(z4.y);
    v.z = y4.z * silu(z4.z);
    v.w = y4.w * silu(z4.w);
    float local = v.x * v.x + v.y * v.y + v.z * v.z + v.w * v.w;

    local = wave_sum(local);
    if ((threadIdx.x & 31) == 0) red[threadIdx.x >> 5] = local;
    __syncthreads();
    if (threadIdx.x == 0) {
        float s = 0.0f;
        #pragma unroll
        for (int i = 0; i < 8; ++i) s += red[i];
        red[0] = rsqrtf(s / (float)DINNER + 1e-5f);
    }
    __syncthreads();
    const float rs = red[0];
    float4 w4 = *(const float4*)(norm_w + i0);
    uint2 o;
    o.x = pack2bf(v.x * rs * w4.x, v.y * rs * w4.y);
    o.y = pack2bf(v.z * rs * w4.z, v.w * rs * w4.w);
    unsigned* orow = (unsigned*)(ybf + (size_t)token * DINNER);
    ((uint2*)orow)[threadIdx.x] = o;
}

// ======================================================================
// ws layout (all 16B aligned):
//   zx   f32 [8192*2192]  71.8 MB
//   xbc  f32 [8192*1152]  37.7 MB
//   yf   f32 [8192*1024]  33.5 MB
//   xbf  bf16[8192*512]    8.4 MB
//   ybf  bf16[8192*1024]  16.8 MB
//   wtin bf16[2240*512]    2.3 MB   (padded, zero-filled)
//   wtout bf16[512*1024]   1.0 MB
// d_in: 0=input, 1..8=params_f, 9..16=params_b
// ======================================================================
extern "C" void kernel_launch(void* const* d_in, const int* in_sizes, int n_in,
                              void* d_out, int out_size, void* d_ws, size_t ws_size,
                              hipStream_t stream) {
    (void)in_sizes; (void)n_in; (void)out_size; (void)ws_size;
    const float* input = (const float*)d_in[0];

    float*  zx    = (float*)d_ws;
    float*  xbc   = zx  + (size_t)NTOK * DPROJ;
    float*  yf    = xbc + (size_t)NTOK * CONVCH;
    __bf16* xbf   = (__bf16*)(yf + (size_t)NTOK * DINNER);
    __bf16* ybf   = xbf + (size_t)NTOK * DMODEL;
    __bf16* wtin  = ybf + (size_t)NTOK * DINNER;
    __bf16* wtout = wtin + (size_t)NPAD_IN * DMODEL;
    float*  out   = (float*)d_out;

    // input -> bf16 (once)
    {
        int n4 = NTOK * DMODEL / 8;
        cvt_f32_bf16_kernel<<<(n4 + 255) / 256, 256, 0, stream>>>(input, (uint4*)xbf, n4);
    }

    for (int dir = 0; dir < 2; ++dir) {
        void* const* P = d_in + 1 + dir * 8;
        const float* inW   = (const float*)P[0];
        const float* convw = (const float*)P[1];
        const float* convb = (const float*)P[2];
        const float* dtb   = (const float*)P[3];
        const float* Alog  = (const float*)P[4];
        const float* Dp    = (const float*)P[5];
        const float* nw    = (const float*)P[6];
        const float* outW  = (const float*)P[7];

        // weights -> transposed bf16 [Npad][K]
        {
            int tot = NPAD_IN * (DMODEL / 2);
            transpose_cvt_kernel<<<(tot + 255) / 256, 256, 0, stream>>>(
                inW, (unsigned*)wtin, DPROJ, NPAD_IN, DMODEL);
            tot = DMODEL * (DINNER / 2);
            transpose_cvt_kernel<<<(tot + 255) / 256, 256, 0, stream>>>(
                outW, (unsigned*)wtout, DMODEL, DMODEL, DINNER);
        }
        // 1) in-proj GEMM  (8192 x 512) @ (512 x 2192)
        {
            dim3 grid(NTOK / BLK_M, NPAD_IN / BLK_N);
            gemm_bf16_async<<<grid, 256, 0, stream>>>(
                xbf, DMODEL, wtin, DMODEL, zx, DPROJ, DPROJ, DMODEL, nullptr, 0);
        }
        // 2) depthwise conv + SiLU
        {
            int n = BATCH * SEQL * CONVCH;
            conv_silu_kernel<<<(n + 255) / 256, 256, 0, stream>>>(
                zx, convw, convb, xbc, dir);
        }
        // 3) selective scan
        ssm_scan_kernel<<<BATCH * NHEADS, 256, 0, stream>>>(
            zx, xbc, dtb, Alog, Dp, yf, dir);
        // 4) gate + RMSNorm -> bf16
        gate_rmsnorm_bf16_kernel<<<NTOK, 256, 0, stream>>>(zx, nw, yf, ybf);
        // 5) out-proj GEMM (8192 x 1024) @ (1024 x 512) + residual
        {
            dim3 grid(NTOK / BLK_M, DMODEL / BLK_N);
            gemm_bf16_async<<<grid, 256, 0, stream>>>(
                ybf, DINNER, wtout, DINNER, out + dir * DMODEL, 2 * DMODEL,
                DMODEL, DINNER, input, DMODEL);
        }
    }
}
